// BidirRecurrentModel_39427799777283
// MI455X (gfx1250) — compile-verified, matrix operand order checked by reading, not run on hardware
//
#include <hip/hip_runtime.h>

// GRU: BS=32, T=2048, D=256, H=256, L=2, bidirectional.
// Phase 1: gx[t][b][l*768+g*256+c] = x @ Wx^T + bx  (bf16, in d_ws, ~192MiB)
//          4-way N-blocked WMMA GEMM: each wave computes a 16x64 strip,
//          A fragment loaded once per k-step and reused across 4 WMMAs.
// Phase 2: 8 blocks = dir(2) x layer(2) x batch-half(2), sequential scan with
//          WMMA bf16 recurrent matmul. r/z Wh fragments in LDS, n Wh in VGPRs.

typedef __attribute__((ext_vector_type(16))) __bf16 v16bf;
typedef __attribute__((ext_vector_type(8)))  float  v8f;

#define T_LEN 2048
#define BSZ   32
#define DDIM  256
#define HDIM  256
#define NCOL  1536   // L*3H
#define G3    768    // 3H
#define RS    264    // padded LDS row stride (bf16 elems): 528B = 132 dwords == 4 mod 64

union FragU { v16bf v; int4 q[2]; };

// Load a 16-lane-wide fragment from a row-major fp32 matrix.
// Caller passes pointer at (row, kbase); layout per ISA 7.12.2:
// elems 0..7 = K kbase+0..7, elems 8..15 = K kbase+16..23.
__device__ inline v16bf load_frag_f32(const float* __restrict__ p) {
  v16bf f;
#pragma unroll
  for (int i = 0; i < 8; ++i)  f[i]     = (__bf16)p[i];
#pragma unroll
  for (int i = 0; i < 8; ++i)  f[8 + i] = (__bf16)p[16 + i];
  return f;
}

// Load a fragment from LDS-resident bf16 row-major data (16B-aligned).
__device__ inline v16bf lds_load_frag(const __bf16* p) {
  FragU u;
  u.q[0] = *(const int4*)(p);
  u.q[1] = *(const int4*)(p + 16);
  return u.v;
}

// Load a pre-formatted raw fragment (32B per lane, contiguous).
__device__ inline v16bf lds_load_raw(const char* base, int lane) {
  FragU u;
  const int4* p = (const int4*)(base + lane * 32);
  u.q[0] = p[0];
  u.q[1] = p[1];
  return u.v;
}

__device__ inline float sigmoid_f(float x) {
  return 1.0f / (1.0f + __expf(-x));
}
__device__ inline float tanh_f(float x) {
  x = fminf(fmaxf(x, -15.0f), 15.0f);
  float e = __expf(2.0f * x);
  return (e - 1.0f) / (e + 1.0f);
}

// ---------------- Phase 1: gx GEMM ----------------
// M = BS*T = 65536 (row m = b*2048 + t, matching x's flat layout),
// N = 1536 (row n = l*768 + g*256 + c, matching Wx's flat layout), K = 256.
// Each wave computes one 16x64 strip (4 N-tiles); 8 waves per block.
__global__ void __launch_bounds__(256) gx_gemm(
    const float* __restrict__ x, const float* __restrict__ Wx,
    const float* __restrict__ bx, __bf16* __restrict__ gx) {
  const int lane = threadIdx.x & 31;
  const int wave = threadIdx.x >> 5;
  const int strip = blockIdx.x * 8 + wave;  // 0 .. 4096*24-1
  const int sn = strip % 24;                // N strip (64 cols)
  const int tm = strip / 24;                // M tile
  const int lrow = lane & 15;
  const int khi  = (lane >> 4) * 8;         // lanes 16..31 take K+8 half

  const int m  = tm * 16 + lrow;            // A row
  const int n0 = sn * 64 + lrow;            // first B row of the strip

  v8f acc[4] = {{}, {}, {}, {}};
#pragma unroll
  for (int kt = 0; kt < 8; ++kt) {
    const int kb = kt * 32 + khi;
    v16bf a = load_frag_f32(x + (long)m * DDIM + kb);
#pragma unroll
    for (int j = 0; j < 4; ++j) {
      v16bf b = load_frag_f32(Wx + (long)(n0 + j * 16) * DDIM + kb);
      acc[j] = __builtin_amdgcn_wmma_f32_16x16x32_bf16(
          false, a, false, b, (short)0, acc[j], false, false);
    }
  }

  // C layout: VGPR i -> M = i (lanes 0-15) / i+8 (lanes 16-31), N = lane&15.
  const int mbase = tm * 16 + ((lane >> 4) * 8);
#pragma unroll
  for (int j = 0; j < 4; ++j) {
    const int ncol = sn * 64 + j * 16 + lrow;
    const float bias = bx[ncol];
#pragma unroll
    for (int i = 0; i < 8; ++i) {
      const int mrow = mbase + i;
      const int t  = mrow & (T_LEN - 1);
      const int bb = mrow >> 11;
      gx[(long)(t * BSZ + bb) * NCOL + ncol] = (__bf16)(acc[j][i] + bias);
    }
  }
}

// ---------------- Phase 2: recurrent scan ----------------
// 8 blocks: dir = bid&1, layer = (bid>>1)&1, batch-half = bid>>2.
// 512 threads = 16 waves; wave w owns hidden columns [16w, 16w+16).
// LDS: [0, 256KB) r/z Wh fragments; then double-buffered h (16 x RS bf16 x 2).
__global__ void __launch_bounds__(512) gru_scan(
    const float* __restrict__ Wh, const float* __restrict__ bh,
    const __bf16* __restrict__ gx, float* __restrict__ out) {
  extern __shared__ char smem[];
  char*   whlds = smem;                                   // 2*16*8*1024 = 262144 B
  __bf16* hbuf0 = (__bf16*)(smem + 262144);               // 16*RS bf16
  __bf16* hbuf1 = hbuf0 + 16 * RS;

  const int dir = blockIdx.x & 1;
  const int l   = (blockIdx.x >> 1) & 1;
  const int bh0 = blockIdx.x >> 2;          // batch half: rows [bh0*16, bh0*16+16)
  const int lane = threadIdx.x & 31;
  const int w    = threadIdx.x >> 5;        // wave id: column chunk
  const int lrow = lane & 15;
  const int khi  = (lane >> 4) * 8;
  const int col  = w * 16 + lrow;           // this lane's hidden column (C layout)

  // --- zero h buffers ---
  {
    int* z = (int*)hbuf0;
    const int nz = (2 * 16 * RS * 2) / 4;   // both buffers, in dwords
    for (int i = threadIdx.x; i < nz; i += 512) z[i] = 0;
  }

  // --- stage Wh fragments: gates r,z -> LDS; gate n -> VGPRs ---
  const float* WhL = Wh + (long)l * G3 * HDIM;
  v16bf whn[8];
#pragma unroll
  for (int kt = 0; kt < 8; ++kt) {
    const int k0 = kt * 32 + khi;
#pragma unroll
    for (int g = 0; g < 2; ++g) {
      v16bf f = load_frag_f32(WhL + (long)(g * HDIM + w * 16 + lrow) * HDIM + k0);
      FragU u; u.v = f;
      int4* d = (int4*)(whlds + ((g * 16 + w) * 8 + kt) * 1024 + lane * 32);
      d[0] = u.q[0];
      d[1] = u.q[1];
    }
    whn[kt] = load_frag_f32(WhL + (long)(2 * HDIM + w * 16 + lrow) * HDIM + k0);
  }

  // Per-lane recurrent biases (column fixed per lane in C layout).
  const float bhr = bh[l * G3 + col];
  const float bhz = bh[l * G3 + 256 + col];
  const float bhn = bh[l * G3 + 512 + col];

  __syncthreads();

  for (int t = 0; t < T_LEN; ++t) {
    const int tt = dir ? (T_LEN - 1 - t) : t;
    __bf16* cur = (t & 1) ? hbuf1 : hbuf0;
    __bf16* nxt = (t & 1) ? hbuf0 : hbuf1;

    // Prefetch input-side gates for this step (hide L2 latency behind WMMA).
    float xr[8], xz[8], xn[8];
#pragma unroll
    for (int i = 0; i < 8; ++i) {
      const int brow  = (lane >> 4) * 8 + i;          // local batch row (C layout)
      const int bglob = bh0 * 16 + brow;
      const long gbase = (long)(tt * BSZ + bglob) * NCOL + l * G3 + col;
      xr[i] = (float)gx[gbase];
      xz[i] = (float)gx[gbase + 256];
      xn[i] = (float)gx[gbase + 512];
    }

    // Recurrent matmul: gh = h @ Wh^T for this wave's 16 columns x 3 gates.
    v8f accr = {}, accz = {}, accn = {};
#pragma unroll
    for (int kt = 0; kt < 8; ++kt) {
      v16bf a  = lds_load_frag(cur + lrow * RS + kt * 32 + khi);
      v16bf br = lds_load_raw(whlds + ((0 * 16 + w) * 8 + kt) * 1024, lane);
      v16bf bz = lds_load_raw(whlds + ((1 * 16 + w) * 8 + kt) * 1024, lane);
      accr = __builtin_amdgcn_wmma_f32_16x16x32_bf16(false, a, false, br, (short)0, accr, false, false);
      accz = __builtin_amdgcn_wmma_f32_16x16x32_bf16(false, a, false, bz, (short)0, accz, false, false);
      accn = __builtin_amdgcn_wmma_f32_16x16x32_bf16(false, a, false, whn[kt], (short)0, accn, false, false);
    }

    // Gate math + state update + output.
#pragma unroll
    for (int i = 0; i < 8; ++i) {
      const int brow  = (lane >> 4) * 8 + i;
      const int bglob = bh0 * 16 + brow;
      const float r  = sigmoid_f(xr[i] + accr[i] + bhr);
      const float z  = sigmoid_f(xz[i] + accz[i] + bhz);
      const float nn = tanh_f(xn[i] + r * (accn[i] + bhn));
      const float hp = (float)cur[brow * RS + col];
      const float hv = (1.0f - z) * nn + z * hp;
      nxt[brow * RS + col] = (__bf16)hv;
      // out: (bs, T*L, 2H); fwd -> [0,H), bwd -> [H,2H)
      out[(long)bglob * (T_LEN * 2 * 512) + (long)(tt * 2 + l) * 512 + dir * 256 + col] = hv;
    }
    __syncthreads();
  }
}

extern "C" void kernel_launch(void* const* d_in, const int* in_sizes, int n_in,
                              void* d_out, int out_size, void* d_ws, size_t ws_size,
                              hipStream_t stream) {
  const float* x  = (const float*)d_in[0];
  const float* Wx = (const float*)d_in[1];
  const float* Wh = (const float*)d_in[2];
  const float* bx = (const float*)d_in[3];
  const float* bh = (const float*)d_in[4];
  float* out = (float*)d_out;
  __bf16* gx = (__bf16*)d_ws;   // needs T*BS*NCOL*2 = 201,326,592 bytes

  // Phase 1: 4096 M-tiles x 24 N-strips, 8 waves (strips) per 256-thread block.
  gx_gemm<<<(4096 * 24) / 8, 256, 0, stream>>>(x, Wx, bx, gx);

  // Phase 2: 8 persistent blocks, 273 KB dynamic LDS each (<= 320 KB/WGP).
  const size_t smem = 2 * 16 * 8 * 1024 + (size_t)2 * 16 * RS * 2;
  gru_scan<<<8, 512, smem, stream>>>(Wh, bh, gx, out);
}
